// Model_61624190763039
// MI455X (gfx1250) — matrix operand order repaired and checked by reading, not compiled, not run on hardware
//
#include <hip/hip_runtime.h>

// dims
#define BB 1024
#define NN 100000
#define DD 256
#define CC 96
#define HH 8
#define NBLK 2
#define NCLS 10
#define NNUM 64
#define DENC 512
#define DFF 1024
#define CHN 8192

typedef __attribute__((ext_vector_type(16))) _Float16 v16h;
typedef __attribute__((ext_vector_type(8)))  _Float16 v8h;
typedef __attribute__((ext_vector_type(4)))  _Float16 v4h;
typedef __attribute__((ext_vector_type(8)))  float    v8f;

#define LDP 40     // padded LDS row stride (halves): 80B, 16B-aligned, bank-spread
#define CTXP 264   // ctx LDS row stride (halves): 528B, 16B-aligned, bank-spread

__device__ __forceinline__ v8h cvt8(float4 a, float4 b) {
  v8h r;
  r[0] = (_Float16)a.x; r[1] = (_Float16)a.y; r[2] = (_Float16)a.z; r[3] = (_Float16)a.w;
  r[4] = (_Float16)b.x; r[5] = (_Float16)b.y; r[6] = (_Float16)b.z; r[7] = (_Float16)b.w;
  return r;
}

// Build a v16h WMMA fragment from two contiguous 16B LDS loads.
__device__ __forceinline__ v16h frag16(const _Float16* p0, const _Float16* p1) {
  v8h lo = *(const v8h*)p0;
  v8h hi = *(const v8h*)p1;
  return __builtin_shufflevector(lo, hi, 0, 1, 2, 3, 4, 5, 6, 7,
                                 8, 9, 10, 11, 12, 13, 14, 15);
}

// CDNA5 async global->LDS copy (16B per lane). LDS byte offset is the low
// 32 bits of the generic pointer.
__device__ __forceinline__ void async_cp16(_Float16* lds, const _Float16* g) {
  unsigned l = (unsigned)(unsigned long long)lds;
  asm volatile("global_load_async_to_lds_b128 %0, %1, off"
               :: "v"(l), "v"(g) : "memory");
}
__device__ __forceinline__ void wait_async0() {
  asm volatile("s_wait_asynccnt 0x0" ::: "memory");
}

// ---------------------------------------------------------------------------
// Tiled GEMM: C[M,N] = act(A[M,K] @ B[K,N] + bias), f32 in/out, f16 WMMA.
// Workgroup tile 64x64; 8 waves (2x4); 2 WMMA per K-step per wave.
// M % 64 == 0, K % 32 == 0, N % 64 == 0.
// ---------------------------------------------------------------------------
__global__ __launch_bounds__(256) void k_gemm(
    const float* __restrict__ A, const float* __restrict__ Bm,
    const float* __restrict__ bias, float* __restrict__ Cm,
    int M, int N, int K, int ldb, int ldc, int act)
{
  __shared__ _Float16 As[64 * LDP];   // [m][k]
  __shared__ _Float16 Bs[64 * LDP];   // transposed: [n][k]
  const int tid  = threadIdx.x;
  const int wave = tid >> 5, lane = tid & 31;
  const int m0 = blockIdx.y * 64;
  const int n0 = blockIdx.x * 64;
  const int wrow = wave >> 2;         // 0..1
  const int wcol = wave & 3;          // 0..3

  v8f acc0 = {}, acc1 = {};

  const int ar  = tid >> 2;           // A row 0..63
  const int ac  = (tid & 3) * 8;      // A k-offset {0,8,16,24}
  const int bkk = tid >> 3;           // B k 0..31
  const int bn8 = (tid & 7) * 8;      // B n-offset

  for (int k0 = 0; k0 < K; k0 += 32) {
    {   // A tile: float4 in, b128 LDS out
      const float* ap = A + (size_t)(m0 + ar) * K + k0 + ac;
      float4 a0 = *(const float4*)ap;
      float4 a1 = *(const float4*)(ap + 4);
      *(v8h*)(As + ar * LDP + ac) = cvt8(a0, a1);
      if (k0 + 32 < K) __builtin_prefetch(ap + 32, 0, 1);
    }
    {   // B tile from [K,N]: coalesced read, transposed 16b LDS writes
      const float* bp = Bm + (size_t)(k0 + bkk) * ldb + n0 + bn8;
      float4 b0 = *(const float4*)bp;
      float4 b1 = *(const float4*)(bp + 4);
      v8h h = cvt8(b0, b1);
#pragma unroll
      for (int j = 0; j < 8; ++j) Bs[(bn8 + j) * LDP + bkk] = h[j];
      if (k0 + 32 < K)
        __builtin_prefetch(Bm + (size_t)(k0 + 32 + bkk) * ldb + n0 + bn8, 0, 1);
    }
    __syncthreads();

    const int kbA = (lane < 16) ? 0 : 8;
    const int kbB = (lane < 16) ? 0 : 8;
    const _Float16* a0p = As + (wrow * 32 + (lane & 15)) * LDP + kbA;
    const _Float16* bp2 = Bs + (wcol * 16 + (lane & 15)) * LDP + kbB;
    v16h af0 = frag16(a0p, a0p + 16);
    v16h af1 = frag16(a0p + 16 * LDP, a0p + 16 * LDP + 16);
    v16h bf  = frag16(bp2, bp2 + 16);
    acc0 = __builtin_amdgcn_wmma_f32_16x16x32_f16(false, af0, false, bf,
                                                  (short)0, acc0, false, false);
    acc1 = __builtin_amdgcn_wmma_f32_16x16x32_f16(false, af1, false, bf,
                                                  (short)0, acc1, false, false);
    __syncthreads();
  }

  const int gn = n0 + wcol * 16 + (lane & 15);
  const float bv = bias ? bias[gn] : 0.f;
  const int mbase = m0 + wrow * 32 + ((lane >> 4) << 3);
#pragma unroll
  for (int r = 0; r < 8; ++r) {
    float v0 = acc0[r] + bv;
    float v1 = acc1[r] + bv;
    if (act == 1) { v0 = v0 > 0.f ? v0 : 0.f; v1 = v1 > 0.f ? v1 : 0.f; }
    Cm[(size_t)(mbase + r) * ldc + gn]      = v0;
    Cm[(size_t)(mbase + 16 + r) * ldc + gn] = v1;
  }
}

// ---------------------------------------------------------------------------
// sims GEMM: C[M,N] = Ah[M,K] @ Bh[N,K]^T, f16 inputs pre-converted.
// Staging via GLOBAL_LOAD_ASYNC_TO_LDS_B128 (ASYNCcnt), double-buffered LDS:
// next tile's async DMA overlaps current tile's WMMA.
// M % 64 == 0, K % 32 == 0; N ragged (reads clamped, ldc covers pad).
// ---------------------------------------------------------------------------
__global__ __launch_bounds__(256) void k_gemm_sims(
    const _Float16* __restrict__ Ah, const _Float16* __restrict__ Bh,
    float* __restrict__ Cm, int M, int N, int K, int ldc)
{
  __shared__ _Float16 As[2][64 * LDP];
  __shared__ _Float16 Bs[2][64 * LDP];
  const int tid  = threadIdx.x;
  const int wave = tid >> 5, lane = tid & 31;
  const int m0 = blockIdx.y * 64;
  const int n0 = blockIdx.x * 64;
  const int wrow = wave >> 2, wcol = wave & 3;

  const int sr = tid >> 2;            // staged row 0..63 (A and B alike)
  const int sk = (tid & 3) * 8;       // k-offset {0,8,16,24}
  int gb = n0 + sr; gb = gb < N ? gb : N - 1;
  const _Float16* agp = Ah + (size_t)(m0 + sr) * K + sk;
  const _Float16* bgp = Bh + (size_t)gb * K + sk;
  _Float16* alp = &As[0][sr * LDP + sk];
  _Float16* blp = &Bs[0][sr * LDP + sk];

  v8f acc0 = {}, acc1 = {};

  async_cp16(alp, agp);               // prologue: tile 0 -> buffer 0
  async_cp16(blp, bgp);
  for (int k0 = 0; k0 < K; k0 += 32) {
    const int buf = (k0 >> 5) & 1;
    wait_async0();                    // own wave's async done ...
    __syncthreads();                  // ... then all waves' -> buffer ready
    if (k0 + 32 < K) {                // kick next tile into the other buffer
      const int nb = (buf ^ 1) * 64 * LDP;
      async_cp16(alp + nb, agp + k0 + 32);
      async_cp16(blp + nb, bgp + k0 + 32);
    }
    const int kbA = (lane < 16) ? 0 : 8;
    const int kbB = (lane < 16) ? 0 : 8;
    const _Float16* a0p = &As[buf][(wrow * 32 + (lane & 15)) * LDP + kbA];
    const _Float16* bp2 = &Bs[buf][(wcol * 16 + (lane & 15)) * LDP + kbB];
    v16h af0 = frag16(a0p, a0p + 16);
    v16h af1 = frag16(a0p + 16 * LDP, a0p + 16 * LDP + 16);
    v16h bf  = frag16(bp2, bp2 + 16);
    acc0 = __builtin_amdgcn_wmma_f32_16x16x32_f16(false, af0, false, bf,
                                                  (short)0, acc0, false, false);
    acc1 = __builtin_amdgcn_wmma_f32_16x16x32_f16(false, af1, false, bf,
                                                  (short)0, acc1, false, false);
  }

  const int gn = n0 + wcol * 16 + (lane & 15);
  const int mbase = m0 + wrow * 32 + ((lane >> 4) << 3);
#pragma unroll
  for (int r = 0; r < 8; ++r) {
    Cm[(size_t)(mbase + r) * ldc + gn]      = acc0[r];
    Cm[(size_t)(mbase + 16 + r) * ldc + gn] = acc1[r];
  }
}

// f32 -> f16 bulk conversion (float4 in, v4h out); n % 4 == 0
__global__ __launch_bounds__(256) void k_f32_to_f16(
    const float* __restrict__ src, _Float16* __restrict__ dst, long n)
{
  long i = ((long)blockIdx.x * 256 + threadIdx.x) * 4;
  if (i < n) {
    float4 v = *(const float4*)(src + i);
    v4h h;
    h[0] = (_Float16)v.x; h[1] = (_Float16)v.y;
    h[2] = (_Float16)v.z; h[3] = (_Float16)v.w;
    *(v4h*)(dst + i) = h;
  }
}

// ---------------------------------------------------------------------------
// Row LayerNorm over D=256 (one block per row)
// ---------------------------------------------------------------------------
__global__ __launch_bounds__(256) void k_layernorm(
    const float* __restrict__ X, const float* __restrict__ s,
    const float* __restrict__ bsh, float* __restrict__ Y)
{
  __shared__ float red[256];
  int b = blockIdx.x, t = threadIdx.x;
  float v = X[b * DD + t];
  red[t] = v;
  __syncthreads();
  for (int off = 128; off > 0; off >>= 1) {
    if (t < off) red[t] += red[t + off];
    __syncthreads();
  }
  float mean = red[0] * (1.f / DD);
  __syncthreads();
  float d = v - mean;
  red[t] = d * d;
  __syncthreads();
  for (int off = 128; off > 0; off >>= 1) {
    if (t < off) red[t] += red[t + off];
    __syncthreads();
  }
  float var = red[0] * (1.f / DD);
  Y[b * DD + t] = d * rsqrtf(var + 1e-5f) * s[t] + bsh[t];
}

__global__ __launch_bounds__(256) void k_l2norm_init(
    const float* __restrict__ Hb, float* __restrict__ Q, float* __restrict__ X)
{
  __shared__ float red[256];
  int b = blockIdx.x, t = threadIdx.x;
  float v = Hb[b * DD + t];
  red[t] = v * v;
  __syncthreads();
  for (int off = 128; off > 0; off >>= 1) {
    if (t < off) red[t] += red[t + off];
    __syncthreads();
  }
  float qv = v * rsqrtf(red[0]);
  Q[b * DD + t] = qv;
  X[b * DD + t] = qv;
}

// ---------------------------------------------------------------------------
// top-k merge over one sims chunk (row stride CHN)
// ---------------------------------------------------------------------------
__global__ __launch_bounds__(256) void k_topk(
    const float* __restrict__ sims, int clen, int cbase,
    float* __restrict__ topv, int* __restrict__ topi, int first)
{
  __shared__ float sv[CHN];
  __shared__ float rvv[CC];
  __shared__ int   rvi[CC];
  __shared__ float redv[256];
  __shared__ int   redl[256];
  __shared__ float outv[CC];
  __shared__ int   outi[CC];
  int b = blockIdx.x, t = threadIdx.x;
  for (int i = t; i < CHN; i += 256)
    sv[i] = (i < clen) ? sims[(size_t)b * CHN + i] : -3.0e38f;
  if (t < CC) {
    rvv[t] = first ? -3.0e38f : topv[b * CC + t];
    rvi[t] = first ? 0 : topi[b * CC + t];
  }
  __syncthreads();
  for (int r = 0; r < CC; ++r) {
    float bv = -3.4e38f;
    int   bl = 0;
    for (int i = t; i < CHN; i += 256)
      if (sv[i] > bv) { bv = sv[i]; bl = i; }
    if (t < CC && rvv[t] > bv) { bv = rvv[t]; bl = CHN + t; }
    redv[t] = bv; redl[t] = bl;
    __syncthreads();
    for (int off = 128; off > 0; off >>= 1) {
      if (t < off && redv[t + off] > redv[t]) {
        redv[t] = redv[t + off]; redl[t] = redl[t + off];
      }
      __syncthreads();
    }
    if (t == 0) {
      int L = redl[0];
      outv[r] = redv[0];
      outi[r] = (L < CHN) ? (cbase + L) : rvi[L - CHN];
      if (L < CHN) sv[L] = -3.4e38f; else rvv[L - CHN] = -3.4e38f;
    }
    __syncthreads();
  }
  if (t < CC) { topv[b * CC + t] = outv[t]; topi[b * CC + t] = outi[t]; }
}

// ---------------------------------------------------------------------------
// Fused cross-attention per query row b. ctx gathered to LDS (f16, padded
// rows), K/V computed with WMMA entirely in LDS, softmax + weighted sum.
// ---------------------------------------------------------------------------
__global__ __launch_bounds__(256) void k_attn(
    const _Float16* __restrict__ ckh, const int* __restrict__ cy,
    const float* __restrict__ lemb, const int* __restrict__ topi,
    const float* __restrict__ qproj,
    const float* __restrict__ wk, const float* __restrict__ bk,
    const float* __restrict__ wv, const float* __restrict__ bv,
    float* __restrict__ obuf)
{
  extern __shared__ char smem[];
  _Float16* ctxh = (_Float16*)smem;            // 96 x CTXP
  _Float16* kh   = ctxh + CC * CTXP;           // 96 x 256
  _Float16* vh   = kh + CC * DD;               // 96 x 256
  _Float16* wt   = vh + CC * DD;               // transposed weight slice: 256 x LDP
  float* qv = (float*)(wt + DD * LDP);         // 256
  float* sc = qv + DD;                         // 8*96 scores

  int b = blockIdx.x, t = threadIdx.x;
  int wave = t >> 5, lane = t & 31;

  // gather context tokens: ckh[idx] + label_emb[y[idx]] (ckh rows L2-resident)
  for (int it = 0; it < 24; ++it) {
    int id = t + 256 * it;                     // 0..6143 4-elem chunks
    int c = id >> 6, d4 = (id & 63) * 4;
    int n = topi[b * CC + c];
    v4h kv4 = *(const v4h*)(ckh + (size_t)n * DD + d4);
    float4 le = *(const float4*)(lemb + (size_t)cy[n] * DD + d4);
    v4h h;
    h[0] = (_Float16)((float)kv4[0] + le.x);
    h[1] = (_Float16)((float)kv4[1] + le.y);
    h[2] = (_Float16)((float)kv4[2] + le.z);
    h[3] = (_Float16)((float)kv4[3] + le.w);
    *(v4h*)(ctxh + c * CTXP + d4) = h;
  }
  qv[t] = qproj[b * DD + t];
  __syncthreads();

  // K and V projections via WMMA: 6x16 = 96 output tiles, 12 per wave
  for (int p = 0; p < 2; ++p) {
    const float* W  = p ? wv : wk;
    const float* Bi = p ? bv : bk;
    _Float16* out = p ? vh : kh;
    v8f acc[12];
#pragma unroll
    for (int tt = 0; tt < 12; ++tt) { v8f z = {}; acc[tt] = z; }
    for (int k0 = 0; k0 < DD; k0 += 32) {
#pragma unroll
      for (int it = 0; it < 8; ++it) {
        int id = t + 256 * it;                 // 0..2047
        int kk = id >> 6, n4 = (id & 63) * 4;
        float4 w4 = *(const float4*)(W + (size_t)(k0 + kk) * DD + n4);
        wt[(n4 + 0) * LDP + kk] = (_Float16)w4.x;
        wt[(n4 + 1) * LDP + kk] = (_Float16)w4.y;
        wt[(n4 + 2) * LDP + kk] = (_Float16)w4.z;
        wt[(n4 + 3) * LDP + kk] = (_Float16)w4.w;
      }
      __syncthreads();
      const int kbA = (lane < 16) ? 0 : 8;
      const int kbB = (lane < 16) ? 0 : 8;
#pragma unroll
      for (int tt = 0; tt < 12; ++tt) {
        int tile = wave + 8 * tt;              // 0..95
        int mt = (tile >> 4) * 16;
        int nt = (tile & 15) * 16;
        const _Float16* ap = ctxh + (mt + (lane & 15)) * CTXP + k0 + kbA;
        const _Float16* bp = wt + (nt + (lane & 15)) * LDP + kbB;
        v16h af = frag16(ap, ap + 16);
        v16h bf = frag16(bp, bp + 16);
        acc[tt] = __builtin_amdgcn_wmma_f32_16x16x32_f16(
            false, af, false, bf, (short)0, acc[tt], false, false);
      }
      __syncthreads();
    }
#pragma unroll
    for (int tt = 0; tt < 12; ++tt) {
      int tile = wave + 8 * tt;
      int mt = (tile >> 4) * 16, nt = (tile & 15) * 16;
      int n = nt + (lane & 15);
      float bb = Bi[n];
#pragma unroll
      for (int r = 0; r < 8; ++r) {
        int m = mt + r + ((lane >> 4) << 3);
        out[m * DD + n] = (_Float16)(acc[tt][r] + bb);
      }
    }
    __syncthreads();
  }

  // scores: 8 heads x 96 ctx, dh = 32
  const float scale = 0.17677669529663687f; // 1/sqrt(32)
#pragma unroll
  for (int j = 0; j < 3; ++j) {
    int pair = t + 256 * j;
    int h = pair / CC, c = pair % CC;
    float s = 0.f;
#pragma unroll
    for (int d = 0; d < 32; ++d)
      s += qv[h * 32 + d] * (float)kh[c * DD + h * 32 + d];
    sc[h * CC + c] = s * scale;
  }
  __syncthreads();

  // softmax: wave w owns head w (wave32)
  {
    int h = wave;
    float x0 = sc[h * CC + lane], x1 = sc[h * CC + lane + 32],
          x2 = sc[h * CC + lane + 64];
    float mx = fmaxf(x0, fmaxf(x1, x2));
    for (int off = 16; off > 0; off >>= 1)
      mx = fmaxf(mx, __shfl_xor(mx, off, 32));
    float e0 = __expf(x0 - mx), e1 = __expf(x1 - mx), e2 = __expf(x2 - mx);
    float sm = e0 + e1 + e2;
    for (int off = 16; off > 0; off >>= 1)
      sm += __shfl_xor(sm, off, 32);
    float inv = 1.f / sm;
    sc[h * CC + lane] = e0 * inv;
    sc[h * CC + lane + 32] = e1 * inv;
    sc[h * CC + lane + 64] = e2 * inv;
  }
  __syncthreads();

  // weighted sum of V
  {
    int h = t >> 5, d = t & 31;
    float o = 0.f;
    for (int c = 0; c < CC; ++c)
      o += sc[h * CC + c] * (float)vh[c * DD + h * 32 + d];
    obuf[b * DD + h * 32 + d] = o;
  }
}

__global__ void k_add(const float* __restrict__ Y, float* __restrict__ X, int n)
{
  int i = blockIdx.x * 256 + threadIdx.x;
  if (i < n) X[i] += Y[i];
}

__global__ __launch_bounds__(256) void k_head(
    const float* __restrict__ X, const float* __restrict__ W,
    const float* __restrict__ Bh, float* __restrict__ Out)
{
  __shared__ float xs[DD];
  int b = blockIdx.x, t = threadIdx.x;
  xs[t] = X[b * DD + t];
  __syncthreads();
  if (t < NCLS) {
    float s = Bh[t];
    for (int d = 0; d < DD; ++d) s += xs[d] * W[d * NCLS + t];
    Out[b * NCLS + t] = s;
  }
}

// ---------------------------------------------------------------------------
extern "C" void kernel_launch(void* const* d_in, const int* in_sizes, int n_in,
                              void* d_out, int out_size, void* d_ws,
                              size_t ws_size, hipStream_t stream)
{
  (void)in_sizes; (void)n_in; (void)out_size; (void)ws_size;
  const float* x_num    = (const float*)d_in[0];
  const float* ck       = (const float*)d_in[1];
  const int*   cy       = (const int*)d_in[2];
  const float* enc_w0   = (const float*)d_in[3];
  const float* enc_b0   = (const float*)d_in[4];
  const float* enc_ln_s = (const float*)d_in[5];
  const float* enc_ln_b = (const float*)d_in[6];
  const float* enc_w1   = (const float*)d_in[7];
  const float* enc_b1   = (const float*)d_in[8];
  const float* enc_w2   = (const float*)d_in[9];
  const float* enc_b2   = (const float*)d_in[10];
  const float* lemb     = (const float*)d_in[11];
  const float* ln1_s    = (const float*)d_in[12];
  const float* ln1_b    = (const float*)d_in[13];
  const float* wq       = (const float*)d_in[14];
  const float* bq       = (const float*)d_in[15];
  const float* wk       = (const float*)d_in[16];
  const float* bk       = (const float*)d_in[17];
  const float* wv       = (const float*)d_in[18];
  const float* bv       = (const float*)d_in[19];
  const float* wo       = (const float*)d_in[20];
  const float* bo       = (const float*)d_in[21];
  const float* ln2_s    = (const float*)d_in[22];
  const float* ln2_b    = (const float*)d_in[23];
  const float* fw1      = (const float*)d_in[24];
  const float* fb1      = (const float*)d_in[25];
  const float* fw2      = (const float*)d_in[26];
  const float* fb2      = (const float*)d_in[27];
  const float* head_w   = (const float*)d_in[28];
  const float* head_b   = (const float*)d_in[29];

  float* ws = (float*)d_ws;
  size_t off = 0;
  auto alloc = [&](size_t n) { float* p = ws + off; off += n; return p; };
  float* h    = alloc((size_t)BB * DD);
  float* hn   = alloc((size_t)BB * DD);
  float* t0   = alloc((size_t)BB * DFF);
  float* q    = alloc((size_t)BB * DD);
  float* x    = alloc((size_t)BB * DD);
  float* xn   = alloc((size_t)BB * DD);
  float* qp   = alloc((size_t)BB * DD);
  float* ob   = alloc((size_t)BB * DD);
  float* o2   = alloc((size_t)BB * DD);
  float* sims = alloc((size_t)BB * CHN);
  float* topv = alloc((size_t)BB * CC);
  int*   topi = (int*)alloc((size_t)BB * CC);
  _Float16* ckh = (_Float16*)alloc((size_t)NN * DD / 2);  // f16 candidate keys
  _Float16* qh  = (_Float16*)alloc((size_t)BB * DD / 2);  // f16 queries

  auto gemm = [&](const float* A, const float* Bm, const float* bias, float* Cm,
                  int M, int Nn, int K, int ldb, int ldc, int act) {
    dim3 g((Nn + 63) / 64, M / 64);
    k_gemm<<<g, 256, 0, stream>>>(A, Bm, bias, Cm, M, Nn, K, ldb, ldc, act);
  };

  // one-off f16 conversion of the candidate bank (amortized over 13 chunks)
  {
    long nck = (long)NN * DD;
    k_f32_to_f16<<<(unsigned)((nck / 4 + 255) / 256), 256, 0, stream>>>(ck, ckh, nck);
  }

  // ---- query encoder ----
  gemm(x_num, enc_w0, enc_b0, h, BB, DD, NNUM, DD, DD, 0);
  for (int i = 0; i < NBLK; ++i) {
    k_layernorm<<<BB, 256, 0, stream>>>(h, enc_ln_s + i * DD, enc_ln_b + i * DD, hn);
    gemm(hn, enc_w1 + (size_t)i * DD * DENC, enc_b1 + i * DENC, t0, BB, DENC, DD, DENC, DENC, 1);
    gemm(t0, enc_w2 + (size_t)i * DENC * DD, enc_b2 + i * DD, h, BB, DD, DENC, DD, DD, 0);
  }
  k_l2norm_init<<<BB, 256, 0, stream>>>(h, q, x);
  {
    long nq = (long)BB * DD;
    k_f32_to_f16<<<(unsigned)((nq / 4 + 255) / 256), 256, 0, stream>>>(q, qh, nq);
  }

  // ---- chunked sims GEMM (async-DMA staged, double-buffered) + top-96 merge ----
  int nch = (NN + CHN - 1) / CHN;
  for (int c = 0; c < nch; ++c) {
    int base = c * CHN;
    int clen = (NN - base < CHN) ? (NN - base) : CHN;
    dim3 g((clen + 63) / 64, BB / 64);
    k_gemm_sims<<<g, 256, 0, stream>>>(qh, ckh + (size_t)base * DD, sims,
                                       BB, clen, DD, CHN);
    k_topk<<<BB, 256, 0, stream>>>(sims, clen, base, topv, topi, c == 0 ? 1 : 0);
  }

  // ---- transformer mixer ----
  size_t attn_smem = (size_t)(CC * CTXP + 2 * CC * DD + DD * LDP) * sizeof(_Float16) +
                     (size_t)(DD + HH * CC) * sizeof(float);
  for (int i = 0; i < NBLK; ++i) {
    k_layernorm<<<BB, 256, 0, stream>>>(x, ln1_s + i * DD, ln1_b + i * DD, xn);
    gemm(xn, wq + (size_t)i * DD * DD, bq + i * DD, qp, BB, DD, DD, DD, DD, 0);
    k_attn<<<BB, 256, attn_smem, stream>>>(ckh, cy, lemb, topi, qp,
        wk + (size_t)i * DD * DD, bk + i * DD,
        wv + (size_t)i * DD * DD, bv + i * DD, ob);
    gemm(ob, wo + (size_t)i * DD * DD, bo + i * DD, o2, BB, DD, DD, DD, DD, 0);
    k_add<<<(BB * DD + 255) / 256, 256, 0, stream>>>(o2, x, BB * DD);
    k_layernorm<<<BB, 256, 0, stream>>>(x, ln2_s + i * DD, ln2_b + i * DD, xn);
    gemm(xn, fw1 + (size_t)i * DD * DFF, fb1 + i * DFF, t0, BB, DFF, DD, DFF, DFF, 1);
    gemm(t0, fw2 + (size_t)i * DFF * DD, fb2 + i * DD, o2, BB, DD, DFF, DD, DD, 0);
    k_add<<<(BB * DD + 255) / 256, 256, 0, stream>>>(o2, x, BB * DD);
  }

  // ---- head ----
  k_head<<<BB, 256, 0, stream>>>(x, head_w, head_b, (float*)d_out);
}
